// EnhancedLearnableMonthlyAggregation_47974784697036
// MI455X (gfx1250) — matrix operand order repaired and checked by reading, not compiled, not run on hardware
//
#include <hip/hip_runtime.h>
#include <hip/hip_bf16.h>
#include <math.h>

// ---------------- problem constants ----------------
static constexpr int cB  = 64;
static constexpr int cND = 1830;
static constexpr int cD  = 512;
static constexpr int cH  = 8;
static constexpr int cHD = 64;
static constexpr int cM  = 60;
static constexpr int cDF = 4 * cD;          // 2048
static constexpr int cRN = cB * cND;        // 117120 rows for K/V projection
static constexpr int cRM = cB * cM;         // 3840 rows for monthly path

// ---------------- WMMA vector types ----------------
typedef __attribute__((ext_vector_type(16))) __bf16          v16bf;
typedef __attribute__((ext_vector_type(8)))  float           v8f;

static __device__ __forceinline__ unsigned short f2bf(float f) {
  return __builtin_bit_cast(unsigned short, (__bf16)f);   // v_cvt, NaN-preserving
}
static __device__ __forceinline__ float bf2f(unsigned short h) {
  return __builtin_bit_cast(float, ((unsigned)h) << 16);
}

// ---------------- small prep kernels ----------------
__global__ __launch_bounds__(256) void build_queries_kernel(
    const float* __restrict__ mq, const float* __restrict__ pos,
    float* __restrict__ q) {
  int i = blockIdx.x * 256 + threadIdx.x;
  if (i < cM * cD) q[i] = mq[i] + pos[i];
}

// W (K x N) fp32 -> WT (N x K) bf16, LDS-tiled (coalesced both directions).
__global__ __launch_bounds__(256) void transpose_cast_kernel(
    const float* __restrict__ W, unsigned short* __restrict__ WT,
    int K, int N) {
  __shared__ float tile[32][33];
  int kb = blockIdx.x * 32, nb = blockIdx.y * 32;
  int tx = threadIdx.x, ty = threadIdx.y;
#pragma unroll
  for (int i = 0; i < 4; ++i) {
    int k = kb + ty + i * 8;
    tile[ty + i * 8][tx] = W[(size_t)k * N + nb + tx];
  }
  __syncthreads();
#pragma unroll
  for (int i = 0; i < 4; ++i) {
    int n = nb + ty + i * 8;
    WT[(size_t)n * K + kb + tx] = f2bf(tile[tx][ty + i * 8]);
  }
}

// per-row mean / rstd for LayerNorm fusion
__global__ __launch_bounds__(256) void row_stats_kernel(
    const float* __restrict__ X, int C,
    float* __restrict__ mu, float* __restrict__ rstd) {
  __shared__ float ss[256], sq[256];
  int r = blockIdx.x;
  float s = 0.f, q = 0.f;
  for (int c = threadIdx.x; c < C; c += 256) {
    float x = X[(size_t)r * C + c];
    s += x; q += x * x;
  }
  ss[threadIdx.x] = s; sq[threadIdx.x] = q;
  __syncthreads();
  for (int st = 128; st; st >>= 1) {
    if (threadIdx.x < st) {
      ss[threadIdx.x] += ss[threadIdx.x + st];
      sq[threadIdx.x] += sq[threadIdx.x + st];
    }
    __syncthreads();
  }
  if (threadIdx.x == 0) {
    float mean = ss[0] / (float)C;
    float var  = sq[0] / (float)C - mean * mean;
    mu[r]   = mean;
    rstd[r] = rsqrtf(var + 1e-5f);
  }
}

// ---------------- templated WMMA GEMM ----------------
// C[R,NSZ] = epilogue( A[R,KSZ] @ WT^T + bias ), KSZ/NSZ compile-time.
//   AMODE: 0 = A fp32 (convert), 1 = A fp32 + fused LayerNorm, 2 = A bf16
//   EPI  : 0 = bias, 1 = bias+GELU(tanh), 2 = bias+residual, 3 = bias->nan0->+residual
// Each wave owns a 16x64 strip: one A fragment feeds 4 WMMAs per k-step.
// Block = 8 waves -> 128x64 block tile. grid = ((R+127)/128, NSZ/64).
// Software-pipelined: fragments for step k+1 are loaded before the WMMAs of
// step k issue; last step peeled so the loop body is branch-free.
// A rows are CLAMPED (not zeroed): no EXEC divergence; garbage rows only
// corrupt C rows that are never stored.
template <int AMODE, int EPI, bool OUTF32, bool OUTBF16, int KSZ, int NSZ>
__global__ __launch_bounds__(256) void wmma_gemm_kernel(
    const void* __restrict__ Av, int R,
    const float* __restrict__ mu, const float* __restrict__ rstd,
    const float* __restrict__ lng, const float* __restrict__ lnb,
    const unsigned short* __restrict__ WTu,
    const float* __restrict__ bias,
    float* __restrict__ Cf, unsigned short* __restrict__ Cbf,
    const float* __restrict__ Res, int resRowMod) {
  const int lane   = threadIdx.x & 31;
  const int wave   = threadIdx.x >> 5;
  const int tileM  = blockIdx.x * 128 + wave * 16;
  const int tileN0 = blockIdx.y * 64;
  const int half   = lane >> 4;
  const int l16    = lane & 15;

  int arow = tileM + l16;
  if (arow >= R) arow = R - 1;               // clamp, store-guarded below
  const __bf16* Abf   = (const __bf16*)Av + (size_t)arow * KSZ;
  const float*  Af32  = (const float*)Av  + (size_t)arow * KSZ;
  const __bf16* bbase = (const __bf16*)WTu + (size_t)(tileN0 + l16) * KSZ;
  const float   m_ = (AMODE == 1) ? mu[arow]   : 0.f;
  const float   r_ = (AMODE == 1) ? rstd[arow] : 0.f;

  auto loadA = [&](int k0) -> v16bf {
    v16bf af;
    if (AMODE == 2) {
      const __bf16* ap = Abf + k0 + half * 8;
#pragma unroll
      for (int i = 0; i < 8; ++i) { af[i] = ap[i]; af[8 + i] = ap[16 + i]; }
    } else {
      const float* ap = Af32 + k0 + half * 8;
      float x[16];
#pragma unroll
      for (int i = 0; i < 8; ++i) { x[i] = ap[i]; x[8 + i] = ap[16 + i]; }
      if (AMODE == 1) {
#pragma unroll
        for (int i = 0; i < 8; ++i) {
          int ka = k0 + half * 8 + i;
          int kb = ka + 16;
          x[i]     = (x[i]     - m_) * r_ * lng[ka] + lnb[ka];
          x[8 + i] = (x[8 + i] - m_) * r_ * lng[kb] + lnb[kb];
        }
      }
#pragma unroll
      for (int i = 0; i < 16; ++i) af[i] = (__bf16)x[i];
    }
    return af;
  };
  auto loadB = [&](int k0, int j) -> v16bf {
    const __bf16* bp = bbase + (size_t)j * 16 * KSZ + k0 + half * 8;
    v16bf b;
#pragma unroll
    for (int i = 0; i < 8; ++i) { b[i] = bp[i]; b[8 + i] = bp[16 + i]; }
    return b;
  };

  v8f acc0 = {}, acc1 = {}, acc2 = {}, acc3 = {};

  // prologue loads
  v16bf nA  = loadA(0);
  v16bf nB0 = loadB(0, 0), nB1 = loadB(0, 1), nB2 = loadB(0, 2), nB3 = loadB(0, 3);

  for (int k0 = 0; k0 < KSZ - 32; k0 += 32) {
    v16bf cA = nA, cB0 = nB0, cB1 = nB1, cB2 = nB2, cB3 = nB3;
    nA  = loadA(k0 + 32);
    nB0 = loadB(k0 + 32, 0); nB1 = loadB(k0 + 32, 1);
    nB2 = loadB(k0 + 32, 2); nB3 = loadB(k0 + 32, 3);
    acc0 = __builtin_amdgcn_wmma_f32_16x16x32_bf16(false, cA, false, cB0, (short)0, acc0, false, false);
    acc1 = __builtin_amdgcn_wmma_f32_16x16x32_bf16(false, cA, false, cB1, (short)0, acc1, false, false);
    acc2 = __builtin_amdgcn_wmma_f32_16x16x32_bf16(false, cA, false, cB2, (short)0, acc2, false, false);
    acc3 = __builtin_amdgcn_wmma_f32_16x16x32_bf16(false, cA, false, cB3, (short)0, acc3, false, false);
  }
  // peeled last step
  acc0 = __builtin_amdgcn_wmma_f32_16x16x32_bf16(false, nA, false, nB0, (short)0, acc0, false, false);
  acc1 = __builtin_amdgcn_wmma_f32_16x16x32_bf16(false, nA, false, nB1, (short)0, acc1, false, false);
  acc2 = __builtin_amdgcn_wmma_f32_16x16x32_bf16(false, nA, false, nB2, (short)0, acc2, false, false);
  acc3 = __builtin_amdgcn_wmma_f32_16x16x32_bf16(false, nA, false, nB3, (short)0, acc3, false, false);

  // ---- epilogue ----
  const int rbase = tileM + half * 8;
  auto epi = [&](bool guard) {
#pragma unroll
    for (int j = 0; j < 4; ++j) {
      const v8f& acc = j == 0 ? acc0 : j == 1 ? acc1 : j == 2 ? acc2 : acc3;
      const int col = tileN0 + j * 16 + l16;
      const float bv = bias[col];
      int rr = 0;
      if (EPI == 2 || EPI == 3) rr = rbase % resRowMod;
      size_t o = (size_t)rbase * NSZ + col;
#pragma unroll
      for (int i = 0; i < 8; ++i, o += NSZ) {
        if (guard && (rbase + i) >= R) break;
        float v = acc[i] + bv;
        if (EPI == 1) {
          float t = 0.7978845608028654f * (v + 0.044715f * v * v * v);
          v = 0.5f * v * (1.f + tanhf(t));
        }
        if (EPI == 3) { if (!(v == v)) v = 0.f; }
        if (EPI == 2 || EPI == 3) {
          v += Res[(size_t)rr * NSZ + col];
          if (++rr == resRowMod) rr = 0;
        }
        if (OUTF32)  Cf[o]  = v;
        if (OUTBF16) Cbf[o] = f2bf(v);
      }
    }
  };
  if (tileM + 16 <= R) epi(false);   // uniform fast path: unguarded stores
  else                 epi(true);    // ragged tail (tiny Q GEMM only)
}

// ---------------- windowed attention (online softmax) ----------------
// grid = B*M blocks, 256 thr = 8 waves = 8 heads. Window length <= ~31.
__global__ __launch_bounds__(256) void attn_kernel(
    const float* __restrict__ Q,            // (M, D) batch-invariant
    const unsigned short* __restrict__ Kb,  // (B, ND, D) bf16
    const unsigned short* __restrict__ Vb,  // (B, ND, D) bf16
    const int* __restrict__ mb,             // (B, M, 2)
    const unsigned char* __restrict__ dmask,// (B, ND)
    float* __restrict__ ctx) {              // (B, M, D)
  int bm = blockIdx.x;
  int b = bm / cM, m = bm % cM;
  int h = threadIdx.x >> 5, lane = threadIdx.x & 31;

  int s = mb[(size_t)bm * 2 + 0];
  int e = mb[(size_t)bm * 2 + 1];
  if (s < 0) s = 0;
  if (e > cND) e = cND;

  const float q0 = Q[(size_t)m * cD + h * cHD + 2 * lane];
  const float q1 = Q[(size_t)m * cD + h * cHD + 2 * lane + 1];

  float mrun = -INFINITY, denom = 0.f, a0 = 0.f, a1 = 0.f;
  for (int n = s; n < e; ++n) {
    if (!dmask[(size_t)b * cND + n]) continue;   // uniform per block
    size_t rowOff = ((size_t)b * cND + n) * cD + h * cHD + 2 * lane;
    unsigned kw = *(const unsigned*)(Kb + rowOff);
    float p = q0 * bf2f((unsigned short)(kw & 0xFFFFu)) +
              q1 * bf2f((unsigned short)(kw >> 16));
    for (int off = 16; off; off >>= 1) p += __shfl_xor(p, off, 32);
    float sc   = p * 0.125f;                     // HD^-0.5 = 1/8
    float mnew = fmaxf(mrun, sc);
    float scl  = __expf(mrun - mnew);            // exp(-inf)=0 first iter
    float w    = __expf(sc - mnew);
    unsigned vw = *(const unsigned*)(Vb + rowOff);
    denom = denom * scl + w;
    a0 = a0 * scl + w * bf2f((unsigned short)(vw & 0xFFFFu));
    a1 = a1 * scl + w * bf2f((unsigned short)(vw >> 16));
    mrun = mnew;
  }
  float* cp = ctx + ((size_t)bm) * cD + h * cHD + 2 * lane;
  if (denom > 0.f) { cp[0] = a0 / denom; cp[1] = a1 / denom; }
  else {
    // all-masked window: reference softmax is NaN; NaN propagates through
    // ctx@Wo and is zeroed by the nan_to_num epilogue.
    float nv = __builtin_nanf("");
    cp[0] = nv; cp[1] = nv;
  }
}

// ---------------- monthly mask ----------------
__global__ __launch_bounds__(256) void month_mask_kernel(
    const int* __restrict__ mb, const unsigned char* __restrict__ dmask,
    float* __restrict__ out) {
  int bm = blockIdx.x * 256 + threadIdx.x;
  if (bm >= cRM) return;
  int b = bm / cM;
  int s = mb[(size_t)bm * 2 + 0];
  int e = mb[(size_t)bm * 2 + 1];
  bool valid = (s < e) && (s < cND) && (e <= cND);
  bool any = false;
  int lo = s < 0 ? 0 : s, hi = e > cND ? cND : e;
  for (int n = lo; n < hi && !any; ++n)
    any = dmask[(size_t)b * cND + n] != 0;
  out[bm] = (valid && any) ? 1.0f : 0.0f;
}

// ---------------- launch ----------------
static size_t alignUp(size_t x) { return (x + 255) & ~(size_t)255; }

extern "C" void kernel_launch(void* const* d_in, const int* in_sizes, int n_in,
                              void* d_out, int out_size, void* d_ws, size_t ws_size,
                              hipStream_t stream) {
  (void)in_sizes; (void)n_in; (void)out_size; (void)ws_size;
  // setup_inputs order
  const float*         daily = (const float*)d_in[0];
  const int*           mb    = (const int*)d_in[1];
  const unsigned char* dmask = (const unsigned char*)d_in[2];
  const float* mq  = (const float*)d_in[3];
  const float* pos = (const float*)d_in[4];
  const float* Wq  = (const float*)d_in[5];  const float* bq  = (const float*)d_in[6];
  const float* Wk  = (const float*)d_in[7];  const float* bk  = (const float*)d_in[8];
  const float* Wv  = (const float*)d_in[9];  const float* bv  = (const float*)d_in[10];
  const float* Wo  = (const float*)d_in[11]; const float* bo  = (const float*)d_in[12];
  const float* g1  = (const float*)d_in[13]; const float* b1n = (const float*)d_in[14];
  const float* g2  = (const float*)d_in[15]; const float* b2n = (const float*)d_in[16];
  const float* W1  = (const float*)d_in[17]; const float* bf1 = (const float*)d_in[18];
  const float* W2  = (const float*)d_in[19]; const float* bf2 = (const float*)d_in[20];
  const float* Wp  = (const float*)d_in[21]; const float* bp  = (const float*)d_in[22];

  // workspace carve-out
  char* ws = (char*)d_ws;
  size_t off = 0;
  auto takeU16 = [&](size_t elems) { unsigned short* p = (unsigned short*)(ws + off); off = alignUp(off + elems * 2); return p; };
  auto takeF32 = [&](size_t elems) { float* p = (float*)(ws + off); off = alignUp(off + elems * 4); return p; };

  unsigned short* WkT = takeU16((size_t)cD * cD);
  unsigned short* WvT = takeU16((size_t)cD * cD);
  unsigned short* WqT = takeU16((size_t)cD * cD);
  unsigned short* WoT = takeU16((size_t)cD * cD);
  unsigned short* WpT = takeU16((size_t)cD * cD);
  unsigned short* W1T = takeU16((size_t)cD * cDF);
  unsigned short* W2T = takeU16((size_t)cDF * cD);
  float* muKV  = takeF32(cRN);  float* rsKV = takeF32(cRN);
  float* muQ   = takeF32(cM);   float* rsQ  = takeF32(cM);
  float* mu2   = takeF32(cRM);  float* rs2  = takeF32(cRM);
  float* quer  = takeF32((size_t)cM * cD);
  float* Qm    = takeF32((size_t)cM * cD);
  unsigned short* Kb = takeU16((size_t)cRN * cD);
  unsigned short* Vb = takeU16((size_t)cRN * cD);
  float* ctx      = takeF32((size_t)cRM * cD);
  float* attended = takeF32((size_t)cRM * cD);
  unsigned short* ffn1 = takeU16((size_t)cRM * cDF);
  float* sum      = takeF32((size_t)cRM * cD);

  float* outH = (float*)d_out;                 // (B, M, D)
  float* outM = outH + (size_t)cRM * cD;       // (B, M) as 0/1 floats

  dim3 blk(256);

  // 1) weights -> bf16 transposed (coalesced LDS-tiled)
  dim3 tblk(32, 8);
  transpose_cast_kernel<<<dim3(cD / 32, cD / 32), tblk, 0, stream>>>(Wk, WkT, cD, cD);
  transpose_cast_kernel<<<dim3(cD / 32, cD / 32), tblk, 0, stream>>>(Wv, WvT, cD, cD);
  transpose_cast_kernel<<<dim3(cD / 32, cD / 32), tblk, 0, stream>>>(Wq, WqT, cD, cD);
  transpose_cast_kernel<<<dim3(cD / 32, cD / 32), tblk, 0, stream>>>(Wo, WoT, cD, cD);
  transpose_cast_kernel<<<dim3(cD / 32, cD / 32), tblk, 0, stream>>>(Wp, WpT, cD, cD);
  transpose_cast_kernel<<<dim3(cD / 32, cDF / 32), tblk, 0, stream>>>(W1, W1T, cD, cDF);
  transpose_cast_kernel<<<dim3(cDF / 32, cD / 32), tblk, 0, stream>>>(W2, W2T, cDF, cD);

  // 2) queries + LN stats
  build_queries_kernel<<<(cM * cD + 255) / 256, blk, 0, stream>>>(mq, pos, quer);
  row_stats_kernel<<<cM, blk, 0, stream>>>(quer, cD, muQ, rsQ);
  row_stats_kernel<<<cRN, blk, 0, stream>>>(daily, cD, muKV, rsKV);

  // 3) big projections (LN fused for K, raw for V) -> bf16
  dim3 gKV((cRN + 127) / 128, cD / 64);
  wmma_gemm_kernel<1, 0, false, true, cD, cD><<<gKV, blk, 0, stream>>>(
      daily, cRN, muKV, rsKV, g1, b1n, WkT, bk, nullptr, Kb, nullptr, 1);
  wmma_gemm_kernel<0, 0, false, true, cD, cD><<<gKV, blk, 0, stream>>>(
      daily, cRN, nullptr, nullptr, nullptr, nullptr, WvT, bv, nullptr, Vb, nullptr, 1);

  // 4) Q (batch-invariant, tiny)
  dim3 gQ(1, cD / 64);
  wmma_gemm_kernel<1, 0, true, false, cD, cD><<<gQ, blk, 0, stream>>>(
      quer, cM, muQ, rsQ, g1, b1n, WqT, bq, Qm, nullptr, nullptr, 1);

  // 5) windowed attention
  attn_kernel<<<cRM, blk, 0, stream>>>(Qm, Kb, Vb, mb, dmask, ctx);

  // 6) attended = queries + nan_to_num(ctx @ Wo + bo)
  dim3 gMN((cRM + 127) / 128, cD / 64);
  wmma_gemm_kernel<0, 3, true, false, cD, cD><<<gMN, blk, 0, stream>>>(
      ctx, cRM, nullptr, nullptr, nullptr, nullptr, WoT, bo,
      attended, nullptr, quer, cM);

  // 7) FFN: LN2 fused into W1 GEMM, GELU epilogue, bf16 intermediate
  row_stats_kernel<<<cRM, blk, 0, stream>>>(attended, cD, mu2, rs2);
  dim3 gF1((cRM + 127) / 128, cDF / 64);
  wmma_gemm_kernel<1, 1, false, true, cD, cDF><<<gF1, blk, 0, stream>>>(
      attended, cRM, mu2, rs2, g2, b2n, W1T, bf1, nullptr, ffn1, nullptr, 1);
  wmma_gemm_kernel<2, 2, true, false, cDF, cD><<<gMN, blk, 0, stream>>>(
      ffn1, cRM, nullptr, nullptr, nullptr, nullptr, W2T, bf2,
      sum, nullptr, attended, cRM);

  // 8) output projection straight into d_out
  wmma_gemm_kernel<0, 0, true, false, cD, cD><<<gMN, blk, 0, stream>>>(
      sum, cRM, nullptr, nullptr, nullptr, nullptr, WpT, bp,
      outH, nullptr, nullptr, 1);

  // 9) monthly mask tail
  month_mask_kernel<<<(cRM + 255) / 256, blk, 0, stream>>>(mb, dmask, outM);
}
